// NeighborhoodPool_1675037245688
// MI455X (gfx1250) — matrix operand order repaired and checked by reading, not compiled, not run on hardware
//
#include <hip/hip_runtime.h>
#include <hip/hip_bf16.h>

typedef __attribute__((ext_vector_type(16))) _Float16 v16h;
typedef __attribute__((ext_vector_type(8)))  _Float16 v8h;
typedef __attribute__((ext_vector_type(8)))  float    v8f;

union H16 { v16h v; v8h h[2]; };

#define N_NODES   100000
#define N_EDGES   1600000
#define DIMF      125
#define DIMH      128
#define NS        8
#define NCLUST    (N_NODES / NS)        // 12500
#define NSORT     131072                // next pow2 >= N_NODES
#define NEG_SLOPE 0.2f
#define LDS_PITCH 136                   // 272B row pitch: 16B-aligned + bank skew

// ---- ordered-uint encoding of float (monotonic increasing) ----
__device__ __forceinline__ unsigned int f2ord(float f) {
    unsigned int b = __float_as_uint(f);
    return (b & 0x80000000u) ? ~b : (b | 0x80000000u);
}
__device__ __forceinline__ float ord2f(unsigned int k) {
    unsigned int b = (k & 0x80000000u) ? (k ^ 0x80000000u) : ~k;
    return __uint_as_float(b);
}

// -------------------------------------------------------------------------
// Kernel 0: init workspace (emax keys to ord(-inf), sums to 0, sort pads)
// -------------------------------------------------------------------------
__global__ void init_ws(unsigned int* __restrict__ emaxKey,
                        float* __restrict__ denom, float* __restrict__ numer,
                        unsigned long long* __restrict__ pairs) {
    int i = blockIdx.x * blockDim.x + threadIdx.x;
    if (i < NSORT) pairs[i] = 0xFFFFFFFFFFFFFFFFull;   // sorts to the end
    if (i < N_NODES) {
        emaxKey[i] = 0x007FFFFFu;   // f2ord(-inf)
        denom[i] = 0.0f;
        numer[i] = 0.0f;
    }
}

// -------------------------------------------------------------------------
// Kernel 1: fused node mat-vecs via WMMA, LDS-staged.
// Block = 256 threads = 8 waves = 128 nodes. shA holds the f16 h-tile
// (concat(x,pos)); shB holds B = [w_src | w_dst | w_lin | 0...] (128x16).
// Per wave: D(16x16,f32) = A(16x128) x B(128x16) in 4 K=32 WMMA steps.
// Fragment layouts per CDNA5 ISA 7.12.2 (wave32):
//   A 16x32 f16 : lane L -> row M = L&15; halves 0..7 = K k0+j (k0=(L>>4)*8),
//                 halves 8..15 = K k0+16+j
//   B 32x16 f16 : lane L -> col N = L&15; same K order as A (k0+j, k0+16+j)
//   C/D 16x16 f32: lane L -> col N = L&15; VGPR v -> row M = (L>>4)*8 + v
// -------------------------------------------------------------------------
__global__ __launch_bounds__(256) void node_matvec_wmma(
    const float* __restrict__ x, const float* __restrict__ pos,
    const float* __restrict__ wsrc, const float* __restrict__ wdst,
    const float* __restrict__ wlin,
    float* __restrict__ xl, float* __restrict__ xr, float* __restrict__ lin,
    int nTiles) {
    __shared__ _Float16 shA[128][LDS_PITCH];
    __shared__ _Float16 shB[16][LDS_PITCH];

    const int tid      = threadIdx.x;
    const int nodeBase = blockIdx.x * 128;

    // ---- branchless cooperative fill: x block is one dense linear range ----
    for (int idx = tid; idx < 128 * DIMF; idx += 256) {
        int n = idx / DIMF, f = idx - n * DIMF;
        int node = nodeBase + n;
        node = (node < N_NODES) ? node : (N_NODES - 1);   // clamp, no branch
        shA[n][f] = (_Float16)x[node * DIMF + f];
    }
    for (int idx = tid; idx < 128 * 3; idx += 256) {      // pos tail cols 125..127
        int n = idx / 3, f = idx - n * 3;
        int node = nodeBase + n;
        node = (node < N_NODES) ? node : (N_NODES - 1);
        shA[n][DIMF + f] = (_Float16)pos[node * 3 + f];
    }
    for (int idx = tid; idx < 16 * DIMH; idx += 256) {    // B matrix, built once
        int ccol = idx >> 7, kk = idx & 127;
        float w = 0.0f;
        if (ccol == 0)      w = wsrc[kk];
        else if (ccol == 1) w = wdst[kk];
        else if (ccol == 2) w = wlin[kk];
        shB[ccol][kk] = (_Float16)w;
    }
    __syncthreads();

    const int wave = tid >> 5;
    const int lane = tid & 31;
    const int tile = blockIdx.x * 8 + wave;
    if (tile >= nTiles) return;                 // wave-uniform; EXEC all-1 inside

    const int row = lane & 15;                  // == D column too
    const int hi  = lane >> 4;
    const int rowLocal = wave * 16 + row;       // node row within shA

    v8f c = {};
    #pragma unroll
    for (int ks = 0; ks < 4; ++ks) {
        const int kb = ks * 32;
        H16 A, B;
        A.h[0] = *(const v8h*)&shA[rowLocal][kb + hi * 8];        // K k0..k0+7
        A.h[1] = *(const v8h*)&shA[rowLocal][kb + hi * 8 + 16];   // K k0+16..+23
        B.h[0] = *(const v8h*)&shB[row][kb + hi * 8];             // K k0..k0+7
        B.h[1] = *(const v8h*)&shB[row][kb + hi * 8 + 16];        // K k0+16..+23
        c = __builtin_amdgcn_wmma_f32_16x16x32_f16(
                false, A.v, false, B.v, (short)0, c, false, false);
    }

    if (row < 3) {                              // scatter D columns 0..2
        float* dstp = (row == 0) ? xl : (row == 1) ? xr : lin;
        #pragma unroll
        for (int v = 0; v < 8; ++v)
            dstp[tile * 16 + hi * 8 + v] = c[v];
    }
}

// -------------------------------------------------------------------------
// Kernel 2/3 helpers: per-edge GATv2 score
// -------------------------------------------------------------------------
__device__ __forceinline__ float edge_score(const int* __restrict__ ei,
                                            const float* __restrict__ pos,
                                            const float* __restrict__ xl,
                                            const float* __restrict__ xr,
                                            const float* __restrict__ wedge,
                                            float attv, int e, int* sOut, int* dOut) {
    int s = ei[e], d = ei[N_EDGES + e];
    float ef = (pos[d * 3 + 0] - pos[s * 3 + 0]) * wedge[0]
             + (pos[d * 3 + 1] - pos[s * 3 + 1]) * wedge[1]
             + (pos[d * 3 + 2] - pos[s * 3 + 2]) * wedge[2];
    float m = xl[s] + xr[d] + ef;
    m = (m > 0.0f) ? m : NEG_SLOPE * m;
    *sOut = s; *dOut = d;
    return m * attv;
}

__global__ void edge_max_kernel(const int* __restrict__ ei,
                                const float* __restrict__ pos,
                                const float* __restrict__ xl,
                                const float* __restrict__ xr,
                                const float* __restrict__ wedge,
                                const float* __restrict__ att,
                                unsigned int* __restrict__ emaxKey) {
    int e = blockIdx.x * blockDim.x + threadIdx.x;
    if (e >= N_EDGES) return;
    int s, d;
    float sc = edge_score(ei, pos, xl, xr, wedge, att[0], e, &s, &d);
    atomicMax(&emaxKey[d], f2ord(sc));
}

__global__ void edge_sum_kernel(const int* __restrict__ ei,
                                const float* __restrict__ pos,
                                const float* __restrict__ xl,
                                const float* __restrict__ xr,
                                const float* __restrict__ wedge,
                                const float* __restrict__ att,
                                const unsigned int* __restrict__ emaxKey,
                                float* __restrict__ denom,
                                float* __restrict__ numer) {
    int e = blockIdx.x * blockDim.x + threadIdx.x;
    if (e >= N_EDGES) return;
    int s, d;
    float sc = edge_score(ei, pos, xl, xr, wedge, att[0], e, &s, &d);
    float ez = __expf(sc - ord2f(emaxKey[d]));
    atomicAdd(&denom[d], ez);
    atomicAdd(&numer[d], ez * xl[s]);
}

// -------------------------------------------------------------------------
// Kernel 4: final per-node score + packed sort keys
// key = (~ord(score))<<32 | idx  -> ascending sort == stable argsort(-score)
// -------------------------------------------------------------------------
__global__ void score_kernel(const float* __restrict__ numer,
                             const float* __restrict__ denom,
                             const float* __restrict__ lin,
                             const float* __restrict__ b_gnn,
                             const float* __restrict__ b_lin,
                             float* __restrict__ score,
                             unsigned long long* __restrict__ pairs) {
    int i = blockIdx.x * blockDim.x + threadIdx.x;
    if (i >= N_NODES) return;
    float sc = numer[i] / (denom[i] + 1e-16f) + b_gnn[0] + lin[i] + b_lin[0];
    score[i] = sc;
    unsigned long long key =
        ((unsigned long long)(~f2ord(sc)) << 32) | (unsigned int)i;
    pairs[i] = key;
}

// -------------------------------------------------------------------------
// Kernel 5: global bitonic sort step (ascending), n = NSORT (pow2)
// -------------------------------------------------------------------------
__global__ void bitonic_step(unsigned long long* __restrict__ d, int j, int k) {
    int i = blockIdx.x * blockDim.x + threadIdx.x;
    int ixj = i ^ j;
    if (ixj > i) {
        unsigned long long a = d[i], b = d[ixj];
        bool up = ((i & k) == 0);
        if ((a > b) == up) { d[i] = b; d[ixj] = a; }
    }
}

// -------------------------------------------------------------------------
// Kernel 6: group-of-8 max pooling + strided pos gather
// -------------------------------------------------------------------------
__global__ __launch_bounds__(128) void pool_kernel(
    const unsigned long long* __restrict__ pairs,
    const float* __restrict__ x, const float* __restrict__ pos,
    float* __restrict__ pooled_x, float* __restrict__ pooled_pos) {
    int c = blockIdx.x;
    int t = threadIdx.x;
    int idx[NS];
    #pragma unroll
    for (int s = 0; s < NS; ++s)
        idx[s] = (int)(unsigned int)(pairs[c * NS + s] & 0xFFFFFFFFull);
    if (t < DIMF) {
        float m = x[idx[0] * DIMF + t];
        #pragma unroll
        for (int s = 1; s < NS; ++s) {
            float v = x[idx[s] * DIMF + t];
            m = (v > m) ? v : m;
        }
        pooled_x[c * DIMF + t] = m;
    }
    if (t < 3)
        pooled_pos[c * 3 + t] = pos[idx[0] * 3 + t];
}

// -------------------------------------------------------------------------
extern "C" void kernel_launch(void* const* d_in, const int* in_sizes, int n_in,
                              void* d_out, int out_size, void* d_ws, size_t ws_size,
                              hipStream_t stream) {
    (void)in_sizes; (void)n_in; (void)out_size; (void)ws_size;
    const float* x     = (const float*)d_in[0];
    const float* pos   = (const float*)d_in[1];
    const float* wsrc  = (const float*)d_in[2];
    const float* wdst  = (const float*)d_in[3];
    const float* wedge = (const float*)d_in[4];
    const float* att   = (const float*)d_in[5];
    const float* bgnn  = (const float*)d_in[6];
    const float* wlin  = (const float*)d_in[7];
    const float* blin  = (const float*)d_in[8];
    const int*   ei    = (const int*)d_in[9];

    // workspace layout (pairs first for 8B alignment)
    char* ws = (char*)d_ws;
    unsigned long long* pairs = (unsigned long long*)ws;        // 1 MB
    float*        xl      = (float*)(ws + (size_t)NSORT * 8);
    float*        xr      = xl + N_NODES;
    float*        lin     = xr + N_NODES;
    unsigned int* emaxKey = (unsigned int*)(lin + N_NODES);
    float*        denom   = (float*)(emaxKey + N_NODES);
    float*        numer   = denom + N_NODES;

    float* out_px = (float*)d_out;                  // 12500 x 125
    float* out_pp = out_px + NCLUST * DIMF;         // 12500 x 3
    float* out_sc = out_pp + NCLUST * 3;            // 100000

    const int nTiles = N_NODES / 16;                // 6250, exact

    init_ws<<<NSORT / 256, 256, 0, stream>>>(emaxKey, denom, numer, pairs);

    node_matvec_wmma<<<(nTiles + 7) / 8, 256, 0, stream>>>(
        x, pos, wsrc, wdst, wlin, xl, xr, lin, nTiles);

    edge_max_kernel<<<N_EDGES / 256, 256, 0, stream>>>(
        ei, pos, xl, xr, wedge, att, emaxKey);

    edge_sum_kernel<<<N_EDGES / 256, 256, 0, stream>>>(
        ei, pos, xl, xr, wedge, att, emaxKey, denom, numer);

    score_kernel<<<(N_NODES + 255) / 256, 256, 0, stream>>>(
        numer, denom, lin, bgnn, blin, out_sc, pairs);

    for (int k = 2; k <= NSORT; k <<= 1)
        for (int j = k >> 1; j > 0; j >>= 1)
            bitonic_step<<<NSORT / 256, 256, 0, stream>>>(pairs, j, k);

    pool_kernel<<<NCLUST, 128, 0, stream>>>(pairs, x, pos, out_px, out_pp);
}
